// SuperGlueMultiHeadAttention_42382737277051
// MI455X (gfx1250) — matrix operand order repaired and checked by reading, not compiled, not run on hardware
//
#include <hip/hip_runtime.h>
#include <hip/hip_bf16.h>

#define FEAT 256
#define HEADS 4
#define HD 64
#define BATCH 4
#define NSEQ 4096

typedef __attribute__((ext_vector_type(16))) __bf16 v16bf;
typedef __attribute__((ext_vector_type(8)))  float  v8f;
typedef __attribute__((ext_vector_type(4)))  unsigned int u32x4;

union ABu { v16bf v; u32x4 q[2]; unsigned u[8]; };

static __device__ __forceinline__ unsigned short f2bf(float f) {
  unsigned x = __float_as_uint(f);
  return (unsigned short)((x + 0x7FFFu + ((x >> 16) & 1u)) >> 16);  // RNE
}

// Assemble a 16-element bf16 WMMA operand from two 16-byte global chunks.
static __device__ __forceinline__ v16bf load2(const unsigned short* p0,
                                              const unsigned short* p1) {
  ABu t;
  t.q[0] = *(const u32x4*)p0;
  t.q[1] = *(const u32x4*)p1;
  return t.v;
}

#define WMMA(A, B, C) \
  __builtin_amdgcn_wmma_f32_16x16x32_bf16(false, (A), false, (B), (short)0, (C), false, false)

// ---------------------------------------------------------------------------
// 1) fp32 weights -> bf16 row-major [o][i], 4 matrices back-to-back
// ---------------------------------------------------------------------------
__global__ void k_cvt_weights(const float* __restrict__ w0, const float* __restrict__ w1,
                              const float* __restrict__ w2, const float* __restrict__ w3,
                              unsigned short* __restrict__ dst) {
  int i = blockIdx.x * blockDim.x + threadIdx.x;          // 0 .. 4*65536-1
  const float* src = (i < 65536) ? w0 : (i < 131072) ? w1 : (i < 196608) ? w2 : w3;
  dst[i] = f2bf(src[i & 65535]);
}

// ---------------------------------------------------------------------------
// 2) (B,256,N) fp32 -> (B,N,256) bf16  (32x32 LDS tile transpose)
// ---------------------------------------------------------------------------
__global__ void k_transpose(const float* __restrict__ src, unsigned short* __restrict__ dst) {
  __shared__ unsigned short t[32][33];
  int tx = threadIdx.x & 31, ty = threadIdx.x >> 5;       // 256 threads
  int n0 = blockIdx.x * 32, c0 = blockIdx.y * 32, b = blockIdx.z;
  const float* sp = src + (size_t)b * FEAT * NSEQ;
  unsigned short* dp = dst + (size_t)b * NSEQ * FEAT;
#pragma unroll
  for (int k = 0; k < 4; ++k) {
    int c = ty + k * 8;
    t[c][tx] = f2bf(sp[(size_t)(c0 + c) * NSEQ + n0 + tx]);
  }
  __syncthreads();
#pragma unroll
  for (int k = 0; k < 4; ++k) {
    int n = ty + k * 8;
    dp[(size_t)(n0 + n) * FEAT + c0 + tx] = t[tx][n];
  }
}

// ---------------------------------------------------------------------------
// 3) Projection GEMM: P[o][n] = W[o][:] . Xt[n][:] + bias[o]
//    mode 0: q -> [bh][n][d], scaled by 1/8   (scale folded in here)
//    mode 1: k -> [bh][n][d]
//    mode 2: v -> [bh][d][n]
// ---------------------------------------------------------------------------
__global__ void __launch_bounds__(256)
k_proj(const unsigned short* __restrict__ Wbf, const float* __restrict__ bias,
       const unsigned short* __restrict__ Xt, unsigned short* __restrict__ dst, int mode) {
  int lane = threadIdx.x & 31, wave = threadIdx.x >> 5;
  int lo = lane & 15, uh = lane >> 4;
  int n0 = (blockIdx.x * 8 + wave) * 16;
  int o0 = blockIdx.y * 16;
  int b  = blockIdx.z;
  const unsigned short* xrow = Xt + ((size_t)b * NSEQ + n0 + lo) * FEAT;
  const unsigned short* wrow = Wbf + (size_t)(o0 + lo) * FEAT;
  v8f acc = {};
#pragma unroll
  for (int cs = 0; cs < 8; ++cs) {
    int cb = cs * 32;
    v16bf A  = load2(wrow + cb + uh * 8,  wrow + cb + uh * 8 + 16);   // A: 16x32 layout
    v16bf Bv = load2(xrow + cb + uh * 8,  xrow + cb + uh * 8 + 16);   // B: 32x16 layout
    acc = WMMA(A, Bv, acc);
  }
  int n = n0 + lo;
#pragma unroll
  for (int r = 0; r < 8; ++r) {
    int o = o0 + r + 8 * uh;                 // channel; C-layout row
    float val = acc[r] + bias[o];
    if (mode == 0) val *= 0.125f;            // 1/sqrt(HEAD_DIM)
    unsigned short bv = f2bf(val);
    int h = o & 3, d = o >> 2;               // c = d*HEADS + h
    if (mode <= 1)
      dst[(((size_t)b * HEADS + h) * NSEQ + n) * HD + d] = bv;
    else
      dst[(((size_t)b * HEADS + h) * HD + d) * NSEQ + n] = bv;
  }
}

// ---------------------------------------------------------------------------
// 4) Flash attention per (batch,head): each wave owns 16 query rows.
//    S^T = K . Q^T so the key index lives on VGPRs -> softmax is in-lane
//    except one shfl_xor(16) per block. O = V . P^T, 4 d-tiles.
// ---------------------------------------------------------------------------
__global__ void __launch_bounds__(256)
k_attn(const unsigned short* __restrict__ Qb, const unsigned short* __restrict__ Kb,
       const unsigned short* __restrict__ Vb, unsigned short* __restrict__ Xa) {
  int lane = threadIdx.x & 31, wave = threadIdx.x >> 5;
  int lo = lane & 15, uh = lane >> 4;
  int bh = blockIdx.y;
  int b = bh >> 2, h = bh & 3;
  int n0 = (blockIdx.x * 8 + wave) * 16;
  const unsigned short* Qh = Qb + (size_t)bh * NSEQ * HD;
  const unsigned short* Kh = Kb + (size_t)bh * NSEQ * HD;
  const unsigned short* Vh = Vb + (size_t)bh * HD * NSEQ;

  // Q as B-operand (loop invariant): 2 d-steps of 32
  const unsigned short* qrow = Qh + (size_t)(n0 + lo) * HD + uh * 8;
  v16bf bq0 = load2(qrow,      qrow + 16);
  v16bf bq1 = load2(qrow + 32, qrow + 48);

  v8f acc0 = {}, acc1 = {}, acc2 = {}, acc3 = {};
  float rmax = -3.0e38f, rsum = 0.0f;

  for (int kb = 0; kb < NSEQ; kb += 32) {
    const unsigned short* k0 = Kh + (size_t)(kb + lo) * HD + uh * 8;
    const unsigned short* k1 = Kh + (size_t)(kb + 16 + lo) * HD + uh * 8;
    v8f st0 = {}, st1 = {};
    st0 = WMMA(load2(k0,      k0 + 16), bq0, st0);   // keys kb..kb+15, d 0..31
    st0 = WMMA(load2(k0 + 32, k0 + 48), bq1, st0);   // d 32..63
    st1 = WMMA(load2(k1,      k1 + 16), bq0, st1);   // keys kb+16..kb+31
    st1 = WMMA(load2(k1 + 32, k1 + 48), bq1, st1);

    // ---- online softmax over the 32 keys (column n = lane%16) ----
    float bm = st0[0];
#pragma unroll
    for (int r = 1; r < 8; ++r) bm = fmaxf(bm, st0[r]);
#pragma unroll
    for (int r = 0; r < 8; ++r) bm = fmaxf(bm, st1[r]);
    bm = fmaxf(bm, __shfl_xor(bm, 16, 32));          // join lane halves
    float nm = fmaxf(rmax, bm);
    float corr = __expf(rmax - nm);
    rmax = nm;
    rsum *= corr;
#pragma unroll
    for (int r = 0; r < 8; ++r) { acc0[r] *= corr; acc1[r] *= corr; acc2[r] *= corr; acc3[r] *= corr; }

    float p[16]; float lsum = 0.0f;
#pragma unroll
    for (int r = 0; r < 8; ++r) { p[r]     = __expf(st0[r] - nm); lsum += p[r]; }
#pragma unroll
    for (int r = 0; r < 8; ++r) { p[8 + r] = __expf(st1[r] - nm); lsum += p[8 + r]; }
    lsum += __shfl_xor(lsum, 16, 32);
    rsum += lsum;

    // ---- C-layout P^T -> B-operand layout (pack only; shuffled qp is unused) ----
    // lower lanes hold m {0-7}(p0-7) {16-23}(p8-15); upper hold {8-15}{24-31}
    unsigned pk[8], qp[8];
#pragma unroll
    for (int j = 0; j < 8; ++j)
      pk[j] = ((unsigned)f2bf(p[2 * j + 1]) << 16) | (unsigned)f2bf(p[2 * j]);
#pragma unroll
    for (int j = 0; j < 8; ++j)
      qp[j] = __shfl_xor(pk[j], 16, 32);
    ABu bp;
#pragma unroll
    for (int j = 0; j < 4; ++j) bp.u[j] = pk[j];
#pragma unroll
    for (int j = 4; j < 8; ++j) bp.u[j] = pk[j];

    // ---- O += V . P^T  (4 d-tiles of 16) ----
    const unsigned short* v0 = Vh + (size_t)lo * NSEQ + kb + uh * 8;
    acc0 = WMMA(load2(v0, v0 + 16), bp.v, acc0);
    const unsigned short* v1 = v0 + (size_t)16 * NSEQ;
    acc1 = WMMA(load2(v1, v1 + 16), bp.v, acc1);
    const unsigned short* v2 = v0 + (size_t)32 * NSEQ;
    acc2 = WMMA(load2(v2, v2 + 16), bp.v, acc2);
    const unsigned short* v3 = v0 + (size_t)48 * NSEQ;
    acc3 = WMMA(load2(v3, v3 + 16), bp.v, acc3);
  }

  float inv = 1.0f / rsum;
  // X_attn[b][n][c] with c = d*4 + h (merge GEMM reads [n][c] contiguously)
  unsigned short* xrow = Xa + ((size_t)b * NSEQ + n0 + lo) * FEAT + h;
#pragma unroll
  for (int r = 0; r < 8; ++r) {
    int dr = r + 8 * uh;
    xrow[(0 * 16 + dr) * 4] = f2bf(acc0[r] * inv);
    xrow[(1 * 16 + dr) * 4] = f2bf(acc1[r] * inv);
    xrow[(2 * 16 + dr) * 4] = f2bf(acc2[r] * inv);
    xrow[(3 * 16 + dr) * 4] = f2bf(acc3[r] * inv);
  }
}

// ---------------------------------------------------------------------------
// 5) Merge GEMM: out[b][o][n] = Wm[o][:] . Xa[b][n][:] + bias[o]  (fp32 out)
// ---------------------------------------------------------------------------
__global__ void __launch_bounds__(256)
k_merge(const unsigned short* __restrict__ Wm, const float* __restrict__ bias,
        const unsigned short* __restrict__ Xa, float* __restrict__ out) {
  int lane = threadIdx.x & 31, wave = threadIdx.x >> 5;
  int lo = lane & 15, uh = lane >> 4;
  int n0 = (blockIdx.x * 8 + wave) * 16;
  int o0 = blockIdx.y * 16;
  int b  = blockIdx.z;
  const unsigned short* xrow = Xa + ((size_t)b * NSEQ + n0 + lo) * FEAT;
  const unsigned short* wrow = Wm + (size_t)(o0 + lo) * FEAT;
  v8f acc = {};
#pragma unroll
  for (int cs = 0; cs < 8; ++cs) {
    int cb = cs * 32;
    v16bf A  = load2(wrow + cb + uh * 8,  wrow + cb + uh * 8 + 16);
    v16bf Bv = load2(xrow + cb + uh * 8,  xrow + cb + uh * 8 + 16);
    acc = WMMA(A, Bv, acc);
  }
  int n = n0 + lo;
#pragma unroll
  for (int r = 0; r < 8; ++r) {
    int o = o0 + r + 8 * uh;
    out[((size_t)b * FEAT + o) * NSEQ + n] = acc[r] + bias[o];
  }
}

// ---------------------------------------------------------------------------
extern "C" void kernel_launch(void* const* d_in, const int* in_sizes, int n_in,
                              void* d_out, int out_size, void* d_ws, size_t ws_size,
                              hipStream_t stream) {
  (void)in_sizes; (void)n_in; (void)out_size; (void)ws_size;
  const float* query = (const float*)d_in[0];
  const float* key   = (const float*)d_in[1];
  const float* value = (const float*)d_in[2];
  const float* w_q   = (const float*)d_in[3];
  const float* b_q   = (const float*)d_in[4];
  const float* w_k   = (const float*)d_in[5];
  const float* b_k   = (const float*)d_in[6];
  const float* w_v   = (const float*)d_in[7];
  const float* b_v   = (const float*)d_in[8];
  const float* w_m   = (const float*)d_in[9];
  const float* b_m   = (const float*)d_in[10];
  float* out = (float*)d_out;

  unsigned short* ws = (unsigned short*)d_ws;
  const size_t XT_E = (size_t)BATCH * NSEQ * FEAT;   // 4M bf16 elements (8 MB)
  const size_t W_E  = (size_t)FEAT * FEAT;           // 65536
  unsigned short* Xt  = ws;                          // reused for q,k,v inputs
  unsigned short* Wbf = Xt + XT_E;                   // 4 matrices
  unsigned short* Qb  = Wbf + 4 * W_E;               // [bh][n][d]
  unsigned short* Kb  = Qb + XT_E;                   // [bh][m][d]
  unsigned short* Vb  = Kb + XT_E;                   // [bh][d][m]
  unsigned short* Xa  = Vb + XT_E;                   // [b][n][c]
  // total ws: ~42.5 MB (fits in 192 MB L2)

  dim3 blk(256);
  dim3 tgrid(NSEQ / 32, FEAT / 32, BATCH);
  dim3 pgrid(NSEQ / 128, FEAT / 16, BATCH);

  k_cvt_weights<<<dim3(4 * 65536 / 256), blk, 0, stream>>>(w_q, w_k, w_v, w_m, Wbf);

  k_transpose<<<tgrid, blk, 0, stream>>>(query, Xt);
  k_proj<<<pgrid, blk, 0, stream>>>(Wbf + 0 * W_E, b_q, Xt, Qb, 0);
  k_transpose<<<tgrid, blk, 0, stream>>>(key, Xt);
  k_proj<<<pgrid, blk, 0, stream>>>(Wbf + 1 * W_E, b_k, Xt, Kb, 1);
  k_transpose<<<tgrid, blk, 0, stream>>>(value, Xt);
  k_proj<<<pgrid, blk, 0, stream>>>(Wbf + 2 * W_E, b_v, Xt, Vb, 2);

  k_attn<<<dim3(NSEQ / 128, BATCH * HEADS), blk, 0, stream>>>(Qb, Kb, Vb, Xa);
  k_merge<<<pgrid, blk, 0, stream>>>(Wbf + 3 * W_E, b_m, Xa, out);
}
